// LowRankRotatedSpaceIntervention_58463094833277
// MI455X (gfx1250) — matrix operand order repaired and checked by reading, not compiled, not run on hardware
//
#include <hip/hip_runtime.h>
#include <hip/hip_bf16.h>

#define EMBED_DIM 4096
#define PROJ_DIM  1024
#define N_PART    8
#define PART_SIZE 128
#define NROWS     8192
#define LDS_K     32
#define LDS_STRIDE 40   // 32 + 8 pad: rows stay 16B-aligned, stride = 20 banks (conflict-free)

typedef __bf16 bf16;
typedef bf16         v16bf __attribute__((ext_vector_type(16)));
typedef float        v8f   __attribute__((ext_vector_type(8)));
typedef unsigned int v8u   __attribute__((ext_vector_type(8)));

// round-to-nearest-even f32 -> bf16
__device__ __forceinline__ unsigned short f2bf(float f) {
  unsigned u = __float_as_uint(f);
  u += 0x7FFFu + ((u >> 16) & 1u);
  return (unsigned short)(u >> 16);
}

// Assemble a 16-element bf16 fragment from two 16-byte LDS chunks.
__device__ __forceinline__ v16bf load_frag(const unsigned short* p0,
                                           const unsigned short* p1) {
  uint4 lo = *(const uint4*)p0;
  uint4 hi = *(const uint4*)p1;
  v8u r;
  r[0] = lo.x; r[1] = lo.y; r[2] = lo.z; r[3] = lo.w;
  r[4] = hi.x; r[5] = hi.y; r[6] = hi.z; r[7] = hi.w;
  return __builtin_bit_cast(v16bf, r);
}

// CDNA5 async copy: 16 bytes per lane, global -> LDS, tracked by ASYNCcnt.
// VDST = per-lane LDS byte offset (low 32 bits of the generic shared address),
// VADDR = per-lane 64-bit global address, SADDR = off (NULL).
__device__ __forceinline__ void async_copy_b128(unsigned lds_off, const void* gptr) {
  asm volatile("global_load_async_to_lds_b128 %0, %1, off"
               :: "v"(lds_off), "v"(gptr)
               : "memory");
}
__device__ __forceinline__ void wait_async() {
  asm volatile("s_wait_asynccnt 0" ::: "memory");
}
__device__ __forceinline__ unsigned lds_addr(const void* p) {
  return (unsigned)(unsigned long long)p;   // LDS offset lives in the low 32 bits
}

// ---------------------------------------------------------------------------
// Kernel 0: convert rotate_weight fp32 [D][P] to bf16 in two layouts:
//   Wbf  [D][P]  (p contiguous)  -> B-tiles of GEMM2
//   WTbf [P][D]  (k contiguous)  -> B-tiles of GEMM1
// ---------------------------------------------------------------------------
__global__ __launch_bounds__(256)
void convert_w_kernel(const float* __restrict__ W,
                      unsigned short* __restrict__ Wbf,
                      unsigned short* __restrict__ WTbf) {
  int i = blockIdx.x * 256 + threadIdx.x;          // 0 .. D*P-1
  int d = i / PROJ_DIM;
  int p = i % PROJ_DIM;
  unsigned short h = f2bf(W[i]);
  Wbf[i] = h;
  WTbf[(size_t)p * EMBED_DIM + d] = h;
}

// ---------------------------------------------------------------------------
// Kernel 1: maskedP[n, p] = select( bf16( (source-base)[n,:] @ W[:,p] ) )
// grid: (NROWS/128, N_PART). Each column tile (128 wide) == one partition,
// so the selection mask is a scalar per row.
// ---------------------------------------------------------------------------
__global__ __launch_bounds__(256)
void gemm1_rotate_mask_kernel(const float* __restrict__ base,
                              const float* __restrict__ source,
                              const int*   __restrict__ subspaces,
                              const unsigned short* __restrict__ WTbf,  // [P][D]
                              unsigned short* __restrict__ maskedP) {   // [N][P] bf16
  __shared__ unsigned short As[2][128][LDS_STRIDE];
  __shared__ unsigned short Bs[2][128][LDS_STRIDE];

  const int tid     = threadIdx.x;
  const int bm      = blockIdx.x;
  const int part    = blockIdx.y;
  const int rowBase = bm * 128;
  const int colBase = part * PART_SIZE;

  const int w     = tid >> 5;
  const int lane  = tid & 31;
  const int half  = lane >> 4;
  const int l16   = lane & 15;
  const int waveM = w & 3;   // 4 row-waves * 32 rows
  const int waveN = w >> 2;  // 2 col-waves * 64 cols

  v8f acc[2][4];
#pragma unroll
  for (int mi = 0; mi < 2; ++mi)
#pragma unroll
    for (int ni = 0; ni < 4; ++ni) acc[mi][ni] = v8f{};

  auto stage = [&](int kt, int buf) {
    // B tile: pure bf16 copy of WTbf rows -> async global->LDS (no VGPR round trip)
#pragma unroll
    for (int it = 0; it < 2; ++it) {
      int idx = tid + it * 256;                 // 0..511
      int r   = idx >> 2;                       // p_local 0..127
      int q   = (idx & 3) << 3;                 // k 0,8,16,24
      async_copy_b128(lds_addr(&Bs[buf][r][q]),
                      WTbf + (size_t)(colBase + r) * EMBED_DIM
                           + (size_t)kt * LDS_K + q);
    }
    // A tile: bf16(source - base), 128 rows x 32 k, convert on the fly (VALU path)
#pragma unroll
    for (int it = 0; it < 4; ++it) {
      int idx = tid + it * 256;                 // 0..1023
      int r   = idx >> 3;                       // row 0..127
      int kq  = (idx & 7) << 2;                 // k 0..28 step 4
      size_t g = (size_t)(rowBase + r) * EMBED_DIM + (size_t)kt * LDS_K + kq;
      float4 b4 = *(const float4*)(base + g);
      float4 s4 = *(const float4*)(source + g);
      unsigned p0 = (unsigned)f2bf(s4.x - b4.x) | ((unsigned)f2bf(s4.y - b4.y) << 16);
      unsigned p1 = (unsigned)f2bf(s4.z - b4.z) | ((unsigned)f2bf(s4.w - b4.w) << 16);
      uint2 pk; pk.x = p0; pk.y = p1;
      *(uint2*)&As[buf][r][kq] = pk;
    }
  };

  const int KT = EMBED_DIM / LDS_K;  // 128
  stage(0, 0);
  wait_async();
  __syncthreads();

  for (int kt = 0; kt < KT; ++kt) {
    int cur = kt & 1;
    if (kt + 1 < KT) stage(kt + 1, cur ^ 1);   // async B + VALU A overlap the WMMAs below

    v16bf afrag[2], bfrag[4];
#pragma unroll
    for (int mi = 0; mi < 2; ++mi) {
      int row = waveM * 32 + mi * 16 + l16;
      afrag[mi] = load_frag(&As[cur][row][half * 8], &As[cur][row][16 + half * 8]);
    }
#pragma unroll
    for (int ni = 0; ni < 4; ++ni) {
      int col = waveN * 64 + ni * 16 + l16;
      bfrag[ni] = load_frag(&Bs[cur][col][half * 8], &Bs[cur][col][16 + half * 8]);
    }
#pragma unroll
    for (int mi = 0; mi < 2; ++mi)
#pragma unroll
      for (int ni = 0; ni < 4; ++ni)
        acc[mi][ni] = __builtin_amdgcn_wmma_f32_16x16x32_bf16(
            false, afrag[mi], false, bfrag[ni], (short)0, acc[mi][ni], false, false);

    wait_async();        // async writes into buffer kt+1 must be LDS-visible
    __syncthreads();     // before any wave reads it next iteration
  }

  // Epilogue: keep this partition only where selected for row n; else write 0.
#pragma unroll
  for (int mi = 0; mi < 2; ++mi) {
#pragma unroll
    for (int r = 0; r < 8; ++r) {
      int n  = rowBase + waveM * 32 + mi * 16 + half * 8 + r;
      int s0 = subspaces[2 * n];
      int s1 = subspaces[2 * n + 1];
      bool keep = (s0 == part) || (s1 == part);
#pragma unroll
      for (int ni = 0; ni < 4; ++ni) {
        int p = colBase + waveN * 64 + ni * 16 + l16;
        maskedP[(size_t)n * PROJ_DIM + p] =
            keep ? f2bf(acc[mi][ni][r]) : (unsigned short)0;
      }
    }
  }
}

// ---------------------------------------------------------------------------
// Kernel 2: out[n, d] = base[n, d] + maskedP[n, :] @ W[d, :]   (K = 1024)
// grid: (EMBED_DIM/128, NROWS/128). Both tiles are pure bf16 copies -> async.
// ---------------------------------------------------------------------------
__global__ __launch_bounds__(256)
void gemm2_unrotate_add_kernel(const float* __restrict__ base,
                               const unsigned short* __restrict__ maskedP, // [N][P]
                               const unsigned short* __restrict__ Wbf,     // [D][P]
                               float* __restrict__ out) {
  __shared__ unsigned short As[2][128][LDS_STRIDE];
  __shared__ unsigned short Bs[2][128][LDS_STRIDE];

  const int tid     = threadIdx.x;
  const int rowBase = blockIdx.y * 128;  // n
  const int colBase = blockIdx.x * 128;  // d

  const int w     = tid >> 5;
  const int lane  = tid & 31;
  const int half  = lane >> 4;
  const int l16   = lane & 15;
  const int waveM = w & 3;
  const int waveN = w >> 2;

  v8f acc[2][4];
#pragma unroll
  for (int mi = 0; mi < 2; ++mi)
#pragma unroll
    for (int ni = 0; ni < 4; ++ni) acc[mi][ni] = v8f{};

  auto stage = [&](int kt, int buf) {
#pragma unroll
    for (int it = 0; it < 2; ++it) {
      int idx = tid + it * 256;  // 0..511
      int r   = idx >> 2;        // 0..127
      int q   = (idx & 3) << 3;  // 0,8,16,24
      async_copy_b128(lds_addr(&As[buf][r][q]),
                      maskedP + (size_t)(rowBase + r) * PROJ_DIM
                              + (size_t)kt * LDS_K + q);
      async_copy_b128(lds_addr(&Bs[buf][r][q]),
                      Wbf + (size_t)(colBase + r) * PROJ_DIM
                          + (size_t)kt * LDS_K + q);
    }
  };

  const int KT = PROJ_DIM / LDS_K;  // 32
  stage(0, 0);
  wait_async();
  __syncthreads();

  for (int kt = 0; kt < KT; ++kt) {
    int cur = kt & 1;
    if (kt + 1 < KT) stage(kt + 1, cur ^ 1);

    v16bf afrag[2], bfrag[4];
#pragma unroll
    for (int mi = 0; mi < 2; ++mi) {
      int row = waveM * 32 + mi * 16 + l16;
      afrag[mi] = load_frag(&As[cur][row][half * 8], &As[cur][row][16 + half * 8]);
    }
#pragma unroll
    for (int ni = 0; ni < 4; ++ni) {
      int col = waveN * 64 + ni * 16 + l16;
      bfrag[ni] = load_frag(&Bs[cur][col][half * 8], &Bs[cur][col][16 + half * 8]);
    }
#pragma unroll
    for (int mi = 0; mi < 2; ++mi)
#pragma unroll
      for (int ni = 0; ni < 4; ++ni)
        acc[mi][ni] = __builtin_amdgcn_wmma_f32_16x16x32_bf16(
            false, afrag[mi], false, bfrag[ni], (short)0, acc[mi][ni], false, false);

    wait_async();
    __syncthreads();
  }

  // Epilogue: residual add with base, fp32 store.
#pragma unroll
  for (int mi = 0; mi < 2; ++mi) {
#pragma unroll
    for (int r = 0; r < 8; ++r) {
      int n = rowBase + waveM * 32 + mi * 16 + half * 8 + r;
#pragma unroll
      for (int ni = 0; ni < 4; ++ni) {
        int d = colBase + waveN * 64 + ni * 16 + l16;
        size_t g = (size_t)n * EMBED_DIM + d;
        out[g] = base[g] + acc[mi][ni][r];
      }
    }
  }
}

// ---------------------------------------------------------------------------
extern "C" void kernel_launch(void* const* d_in, const int* in_sizes, int n_in,
                              void* d_out, int out_size, void* d_ws, size_t ws_size,
                              hipStream_t stream) {
  const float* base      = (const float*)d_in[0];
  const float* source    = (const float*)d_in[1];
  const int*   subspaces = (const int*)d_in[2];
  const float* W         = (const float*)d_in[3];
  float*       out       = (float*)d_out;

  // workspace layout (bf16 halves): Wbf [D][P] | WTbf [P][D] | maskedP [N][P]
  unsigned short* Wbf     = (unsigned short*)d_ws;
  unsigned short* WTbf    = Wbf  + (size_t)EMBED_DIM * PROJ_DIM;
  unsigned short* maskedP = WTbf + (size_t)EMBED_DIM * PROJ_DIM;

  convert_w_kernel<<<(EMBED_DIM * PROJ_DIM) / 256, 256, 0, stream>>>(W, Wbf, WTbf);

  gemm1_rotate_mask_kernel<<<dim3(NROWS / 128, N_PART), 256, 0, stream>>>(
      base, source, subspaces, WTbf, maskedP);

  gemm2_unrotate_add_kernel<<<dim3(EMBED_DIM / 128, NROWS / 128), 256, 0, stream>>>(
      base, maskedP, Wbf, out);
}
